// TransformerBlock_55113020342627
// MI455X (gfx1250) — matrix operand order repaired and checked by reading, not compiled, not run on hardware
//
#include <hip/hip_runtime.h>
#include <hip/hip_bf16.h>

// ---------------------------------------------------------------------------
// Transformer block for MI455X (gfx1250, wave32, WMMA).
// B=2, T=2048, D=1024, H=16, HD=64.  All GEMMs + attention use
// v_wmma_f32_16x16x32_f16 (f16 in / f32 accumulate).  GEMM tiles are staged
// with GLOBAL_LOAD_ASYNC_TO_LDS_B128 (ASYNCcnt) and double-buffered LDS so
// HBM traffic overlaps the WMMA stream.
// Workspace requirement: ~101 MB.
// ---------------------------------------------------------------------------

#define DD    1024
#define HH    16
#define HDIM  64
#define TT    2048
#define BBATCH 2
#define MTOT  (BBATCH * TT)      // 4096 rows

typedef _Float16 f16;
typedef __attribute__((ext_vector_type(8)))  _Float16 v8h;
typedef __attribute__((ext_vector_type(16))) _Float16 v16h;
typedef __attribute__((ext_vector_type(8)))  float    v8f;
typedef __attribute__((ext_vector_type(4)))  int      v4i;

static __device__ __forceinline__ v8h ld8(const f16* p) { return *(const v8h*)p; }

static __device__ __forceinline__ v16h cat16(v8h lo, v8h hi) {
    union { v16h w; v8h h[2]; } u;
    u.h[0] = lo; u.h[1] = hi;
    return u.w;
}

static __device__ __forceinline__ v8f vzero8() {
    v8f z = {0.f, 0.f, 0.f, 0.f, 0.f, 0.f, 0.f, 0.f};
    return z;
}

static __device__ __forceinline__ v8f wmma_f16(v16h a, v16h b, v8f c) {
    // D = A(16x32 f16) * B(32x16 f16) + C(16x16 f32)
    return __builtin_amdgcn_wmma_f32_16x16x32_f16(false, a, false, b,
                                                  (short)0, c, false, false);
}

// ---- CDNA5 async global->LDS copy (ASYNCcnt-tracked), 16 bytes/lane -------
static __device__ __forceinline__ void async_copy_b128(void* lds, const f16* g) {
#if __has_builtin(__builtin_amdgcn_global_load_async_to_lds_b128)
    __builtin_amdgcn_global_load_async_to_lds_b128(
        (__attribute__((address_space(1))) v4i*)(uintptr_t)g,
        (__attribute__((address_space(3))) v4i*)(uintptr_t)lds,
        0, 0);
#else
    asm volatile("global_load_async_to_lds_b128 %0, %1, off"
                 :: "v"((unsigned int)(uintptr_t)lds),
                    "v"((unsigned long long)(uintptr_t)g)
                 : "memory");
#endif
}

static __device__ __forceinline__ void wait_async0() {
#if __has_builtin(__builtin_amdgcn_s_wait_asynccnt)
    __builtin_amdgcn_s_wait_asynccnt(0);
#else
    asm volatile("s_wait_asynccnt 0x0" ::: "memory");
#endif
}

// ---------------------------------------------------------------------------
// Weight transpose + convert: src fp32 [K x N] row-major -> dst f16 [N x K].
// Gives both WMMA operands contiguous-K fragment loads.
// ---------------------------------------------------------------------------
__global__ __launch_bounds__(256) void k_transpose_f32_f16(
    const float* __restrict__ src, f16* __restrict__ dst, int K, int N)
{
    __shared__ float tile[32][33];
    const int n0 = blockIdx.x * 32, k0 = blockIdx.y * 32;
    const int tx = threadIdx.x, ty = threadIdx.y;       // block (32, 8)
    #pragma unroll
    for (int j = 0; j < 4; ++j)
        tile[ty + 8 * j][tx] = src[(size_t)(k0 + ty + 8 * j) * N + n0 + tx];
    __syncthreads();
    #pragma unroll
    for (int j = 0; j < 4; ++j)
        dst[(size_t)(n0 + ty + 8 * j) * K + k0 + tx] = (f16)tile[tx][ty + 8 * j];
}

// ---------------------------------------------------------------------------
// LayerNorm over D=1024, one row per 256-thread block, f16 output.
// ---------------------------------------------------------------------------
__global__ __launch_bounds__(256) void k_layernorm_f16(
    const float* __restrict__ x, const float* __restrict__ gamma,
    const float* __restrict__ beta, f16* __restrict__ out)
{
    const int row = blockIdx.x;
    const int tid = threadIdx.x;
    const float4 v = ((const float4*)(x + (size_t)row * DD))[tid];
    float a[4] = {v.x, v.y, v.z, v.w};
    float s = 0.f, s2 = 0.f;
    #pragma unroll
    for (int j = 0; j < 4; ++j) { s += a[j]; s2 += a[j] * a[j]; }
    #pragma unroll
    for (int m = 16; m >= 1; m >>= 1) {
        s  += __shfl_xor(s,  m, 32);
        s2 += __shfl_xor(s2, m, 32);
    }
    __shared__ float ps[8], ps2[8];
    if ((tid & 31) == 0) { ps[tid >> 5] = s; ps2[tid >> 5] = s2; }
    __syncthreads();
    s = 0.f; s2 = 0.f;
    #pragma unroll
    for (int w = 0; w < 8; ++w) { s += ps[w]; s2 += ps2[w]; }
    const float mu   = s * (1.0f / DD);
    const float var  = s2 * (1.0f / DD) - mu * mu;
    const float rstd = rsqrtf(var + 1e-5f);
    f16* orow = out + (size_t)row * DD;
    #pragma unroll
    for (int j = 0; j < 4; ++j) {
        const int c = tid * 4 + j;
        orow[c] = (f16)((a[j] - mu) * rstd * gamma[c] + beta[c]);
    }
}

// ---------------------------------------------------------------------------
// Tiled WMMA GEMM:  C[M x N] = A[M x K] * Bt[N x K]^T  (both f16, f32 acc)
// Block tile 128x128, BK=32, 8 waves, each wave 2x4 WMMA accumulators.
// Double-buffered LDS; tiles staged by GLOBAL_LOAD_ASYNC_TO_LDS_B128 so the
// next chunk's HBM traffic overlaps the current chunk's 8 WMMAs.
// Epilogue modes:
//   0: scatter f16 to QKV [3][B,H,T,HD] layout      (outH = qkv base)
//   1: f32 out = acc + resid                         (Wo -> x1)
//   2: f16 out = gelu(acc + bias)                    (W1 -> ffn act)
//   3: f32 out = acc + bias + resid                  (W2 -> d_out)
// ---------------------------------------------------------------------------
#define BM  128
#define BN  128
#define BKC 32
#define LST 40   // LDS row stride in halfs (32 + 8 pad; 80B keeps 16B alignment)

__global__ __launch_bounds__(256) void k_gemm_f16_wmma(
    const f16* __restrict__ A, const f16* __restrict__ Bt,
    int M, int N, int K, int mode,
    const float* __restrict__ bias, const float* __restrict__ resid,
    float* __restrict__ outF, f16* __restrict__ outH)
{
    __shared__ __attribute__((aligned(16))) f16 As[2][BM][LST];  // 2 x 10240 B
    __shared__ __attribute__((aligned(16))) f16 Bs[2][BN][LST];  // 2 x 10240 B

    const int tid  = threadIdx.x;
    const int wid  = tid >> 5, lane = tid & 31;
    const int l16  = lane & 15, hlf = lane >> 4;
    const int kb   = hlf * 8;
    const int bm   = blockIdx.y * BM, bn = blockIdx.x * BN;
    const int wm   = (wid & 3) * 32, wn = (wid >> 2) * 64;

    v8f acc[2][4];
    #pragma unroll
    for (int tm = 0; tm < 2; ++tm)
        #pragma unroll
        for (int tn = 0; tn < 4; ++tn) acc[tm][tn] = vzero8();

    const int r = tid >> 1;            // 0..127: tile row
    const int c = (tid & 1) * 16;      // 0 or 16: half-chunk of BK
    const f16* ag = A  + (size_t)(bm + r) * K + c;
    const f16* bg = Bt + (size_t)(bn + r) * K + c;

    // prologue: stage chunk 0 into buffer 0 (async, no VGPR staging)
    async_copy_b128(&As[0][r][c],     ag);
    async_copy_b128(&As[0][r][c + 8], ag + 8);
    async_copy_b128(&Bs[0][r][c],     bg);
    async_copy_b128(&Bs[0][r][c + 8], bg + 8);

    const int nIter = K / BKC;
    int buf = 0;
    for (int i = 0; i < nIter; ++i) {
        wait_async0();                 // our chunk-i copies have landed in LDS
        __syncthreads();               // everyone's chunk i landed; prev reads done
        if (i + 1 < nIter) {           // issue chunk i+1 into the other buffer
            const f16* ag2 = ag + (size_t)(i + 1) * BKC;
            const f16* bg2 = bg + (size_t)(i + 1) * BKC;
            async_copy_b128(&As[buf ^ 1][r][c],     ag2);
            async_copy_b128(&As[buf ^ 1][r][c + 8], ag2 + 8);
            async_copy_b128(&Bs[buf ^ 1][r][c],     bg2);
            async_copy_b128(&Bs[buf ^ 1][r][c + 8], bg2 + 8);
        }

        v16h af[2], bf[4];
        #pragma unroll
        for (int tm = 0; tm < 2; ++tm) {
            const f16* p = &As[buf][wm + tm * 16 + l16][0];
            af[tm] = cat16(ld8(p + kb), ld8(p + 16 + kb));            // A frag layout
        }
        #pragma unroll
        for (int tn = 0; tn < 4; ++tn) {
            const f16* p = &Bs[buf][wn + tn * 16 + l16][0];
            bf[tn] = cat16(ld8(p + kb), ld8(p + 16 + kb)); // B frag layout
        }
        #pragma unroll
        for (int tm = 0; tm < 2; ++tm)
            #pragma unroll
            for (int tn = 0; tn < 4; ++tn)
                acc[tm][tn] = wmma_f16(af[tm], bf[tn], acc[tm][tn]);
        buf ^= 1;
    }

    #pragma unroll
    for (int tm = 0; tm < 2; ++tm)
        #pragma unroll
        for (int tn = 0; tn < 4; ++tn)
            #pragma unroll
            for (int g = 0; g < 8; ++g) {
                const int m = bm + wm + tm * 16 + g + 8 * hlf;
                const int n = bn + wn + tn * 16 + l16;
                const float val = acc[tm][tn][g];
                if (mode == 0) {
                    const int which = n >> 10, nn = n & 1023;
                    const int head = nn >> 6, hd = nn & 63;
                    const int bb = m >> 11, t = m & (TT - 1);
                    outH[(size_t)which * (BBATCH * TT * DD) +
                         ((size_t)(bb * HH + head) * TT + t) * HDIM + hd] = (f16)val;
                } else if (mode == 1) {
                    outF[(size_t)m * N + n] = val + resid[(size_t)m * N + n];
                } else if (mode == 2) {
                    const float u = val + bias[n];
                    const float gl = 0.5f * u * (1.0f + erff(u * 0.70710678118654752f));
                    outH[(size_t)m * N + n] = (f16)gl;
                } else {
                    outF[(size_t)m * N + n] = val + bias[n] + resid[(size_t)m * N + n];
                }
            }
}

// ---------------------------------------------------------------------------
// Causal flash attention, one (b, h, 128-row q-tile) per 256-thread block.
// Each wave owns 16 q rows; all 8 waves cooperatively stage 32-key K/V chunks
// (V transposed in LDS so P*V B-fragments are contiguous-K).
// S = Q*K^T and O += P*V via v_wmma_f32_16x16x32_f16; online softmax in f32.
// ---------------------------------------------------------------------------
__global__ __launch_bounds__(256) void k_flash_attn(
    const f16* __restrict__ q, const f16* __restrict__ k,
    const f16* __restrict__ v, f16* __restrict__ ctxo)
{
    __shared__ __attribute__((aligned(16))) f16 Ks[32][72];      // [key][hd]
    __shared__ __attribute__((aligned(16))) f16 Vs[64][40];      // [hd][key] (transposed)
    __shared__ __attribute__((aligned(16))) f16 Ps[8][16][40];   // per-wave P tile

    const int bb = blockIdx.z, hh = blockIdx.y, qt = blockIdx.x;
    const int tid = threadIdx.x, wid = tid >> 5, lane = tid & 31;
    const int l16 = lane & 15, hlf = lane >> 4, kb = hlf * 8;

    const size_t base = (size_t)(bb * HH + hh) * TT * HDIM;
    const f16* qh = q + base;
    const f16* kh = k + base;
    const f16* vh = v + base;

    const int qbase = qt * 128 + wid * 16;

    // Q fragments for this wave's 16 rows (K-dim = HD = 64 -> two A frags)
    const f16* qrow = qh + (size_t)(qbase + l16) * HDIM;
    const v16h Aq0 = cat16(ld8(qrow + kb),      ld8(qrow + 16 + kb));
    const v16h Aq1 = cat16(ld8(qrow + 32 + kb), ld8(qrow + 48 + kb));

    v8f O[4];
    #pragma unroll
    for (int nt = 0; nt < 4; ++nt) O[nt] = vzero8();
    float mrun[8], lrun[8];
    #pragma unroll
    for (int g = 0; g < 8; ++g) { mrun[g] = -1e30f; lrun[g] = 0.f; }

    const int nchunks = qt * 4 + 4;        // causal: keys 0 .. qt*128+127
    const int ldr = tid >> 3;              // 0..31 key row within chunk
    const int ldc = (tid & 7) * 8;         // 0..56 hd column base

    for (int cidx = 0; cidx < nchunks; ++cidx) {
        const int kc = cidx * 32;
        __syncthreads();
        // cooperative staging of K (row-major, async direct-to-LDS) and
        // V (transposed, via VGPR round trip)
        {
            async_copy_b128(&Ks[ldr][ldc], kh + (size_t)(kc + ldr) * HDIM + ldc);
            const v8h vd = ld8(vh + (size_t)(kc + ldr) * HDIM + ldc);
            #pragma unroll
            for (int e = 0; e < 8; ++e) Vs[ldc + e][ldr] = vd[e];
            wait_async0();
        }
        __syncthreads();

        if (kc <= qbase + 15) {            // chunk has unmasked keys for this wave
            // ---- S = Q * K^T (two 16x16 tiles covering 32 keys) ----
            v8f S0 = vzero8(), S1 = vzero8();
            {
                const f16* kp0 = &Ks[l16][0];
                S0 = wmma_f16(Aq0, cat16(ld8(kp0 + kb),      ld8(kp0 + 16 + kb)),      S0);
                S0 = wmma_f16(Aq1, cat16(ld8(kp0 + 32 + kb), ld8(kp0 + 48 + kb)), S0);
                const f16* kp1 = &Ks[16 + l16][0];
                S1 = wmma_f16(Aq0, cat16(ld8(kp1 + kb),      ld8(kp1 + 16 + kb)),      S1);
                S1 = wmma_f16(Aq1, cat16(ld8(kp1 + 32 + kb), ld8(kp1 + 48 + kb)), S1);
            }
            // ---- online softmax (scale 1/sqrt(64) = 0.125) ----
            const int ki0 = kc + l16, ki1 = kc + 16 + l16;
            float p0[8], p1[8];
            #pragma unroll
            for (int g = 0; g < 8; ++g) {
                const int qi = qbase + g + 8 * hlf;
                const float s0 = (ki0 <= qi) ? S0[g] * 0.125f : -1e30f;
                const float s1 = (ki1 <= qi) ? S1[g] * 0.125f : -1e30f;
                float rmax = fmaxf(s0, s1);
                rmax = fmaxf(rmax, __shfl_xor(rmax, 1, 32));
                rmax = fmaxf(rmax, __shfl_xor(rmax, 2, 32));
                rmax = fmaxf(rmax, __shfl_xor(rmax, 4, 32));
                rmax = fmaxf(rmax, __shfl_xor(rmax, 8, 32));
                const float mnew = fmaxf(mrun[g], rmax);
                const float corr = __expf(mrun[g] - mnew);
                mrun[g] = mnew;
                p0[g] = __expf(s0 - mnew);
                p1[g] = __expf(s1 - mnew);
                float rsum = p0[g] + p1[g];
                rsum += __shfl_xor(rsum, 1, 32);
                rsum += __shfl_xor(rsum, 2, 32);
                rsum += __shfl_xor(rsum, 4, 32);
                rsum += __shfl_xor(rsum, 8, 32);
                lrun[g] = lrun[g] * corr + rsum;
                #pragma unroll
                for (int nt = 0; nt < 4; ++nt) O[nt][g] *= corr;
            }
            // ---- P (16x32) to per-wave LDS, reload in A-fragment layout ----
            #pragma unroll
            for (int g = 0; g < 8; ++g) {
                Ps[wid][g + 8 * hlf][l16]      = (f16)p0[g];
                Ps[wid][g + 8 * hlf][16 + l16] = (f16)p1[g];
            }
            asm volatile("s_wait_dscnt 0" ::: "memory");   // same-wave LDS RAW
            const f16* pp = &Ps[wid][l16][0];
            const v16h Pf = cat16(ld8(pp + kb), ld8(pp + 16 + kb));
            // ---- O += P * V ----
            #pragma unroll
            for (int nt = 0; nt < 4; ++nt) {
                const f16* vp = &Vs[nt * 16 + l16][0];
                const v16h Vf = cat16(ld8(vp + kb), ld8(vp + 16 + kb));
                O[nt] = wmma_f16(Pf, Vf, O[nt]);
            }
        }
    }

    // ---- normalize and scatter to ctx [B, T, D] ----
    #pragma unroll
    for (int nt = 0; nt < 4; ++nt)
        #pragma unroll
        for (int g = 0; g < 8; ++g) {
            const int t = qbase + g + 8 * hlf;
            const float val = O[nt][g] / lrun[g];
            ctxo[(size_t)(bb * TT + t) * DD + hh * HDIM + nt * 16 + l16] = (f16)val;
        }
}

// ---------------------------------------------------------------------------
// Host-side orchestration (all launches on `stream`; no allocation).
// ---------------------------------------------------------------------------
extern "C" void kernel_launch(void* const* d_in, const int* in_sizes, int n_in,
                              void* d_out, int out_size, void* d_ws, size_t ws_size,
                              hipStream_t stream)
{
    (void)in_sizes; (void)n_in; (void)out_size; (void)ws_size;

    const float* x    = (const float*)d_in[0];
    const float* ln1g = (const float*)d_in[1];
    const float* ln1b = (const float*)d_in[2];
    const float* Wq   = (const float*)d_in[3];
    const float* Wk   = (const float*)d_in[4];
    const float* Wv   = (const float*)d_in[5];
    const float* Wo   = (const float*)d_in[6];
    const float* ln2g = (const float*)d_in[7];
    const float* ln2b = (const float*)d_in[8];
    const float* W1   = (const float*)d_in[9];
    const float* b1   = (const float*)d_in[10];
    const float* W2   = (const float*)d_in[11];
    const float* b2   = (const float*)d_in[12];
    float* out = (float*)d_out;

    // workspace carve (256B-aligned slices); total ~101 MB
    char* ws = (char*)d_ws;
    size_t off = 0;
    auto carve = [&](size_t bytes) -> void* {
        void* p = ws + off;
        off += (bytes + 255) & ~(size_t)255;
        return p;
    };
    f16*  wt_qkv = (f16*)carve((size_t)3 * DD * DD * sizeof(f16)); // [3N=3072][K=1024]
    f16*  wt_o   = (f16*)carve((size_t)DD * DD * sizeof(f16));     // [1024][1024]
    f16*  wt_1   = (f16*)carve((size_t)2 * DD * DD * sizeof(f16)); // [2048][1024]
    f16*  wt_2   = (f16*)carve((size_t)2 * DD * DD * sizeof(f16)); // [1024][2048]
    f16*  h1     = (f16*)carve((size_t)MTOT * DD * sizeof(f16));
    f16*  qkv    = (f16*)carve((size_t)3 * MTOT * DD * sizeof(f16));
    f16*  ctxb   = (f16*)carve((size_t)MTOT * DD * sizeof(f16));
    float* x1    = (float*)carve((size_t)MTOT * DD * sizeof(float));
    f16*  h2     = (f16*)carve((size_t)MTOT * DD * sizeof(f16));
    f16*  ffn    = (f16*)carve((size_t)MTOT * 2 * DD * sizeof(f16));

    const dim3 tpb_t(32, 8);
    // weight transposes to f16 [N x K]
    k_transpose_f32_f16<<<dim3(DD / 32, DD / 32), tpb_t, 0, stream>>>(Wq, wt_qkv,                DD, DD);
    k_transpose_f32_f16<<<dim3(DD / 32, DD / 32), tpb_t, 0, stream>>>(Wk, wt_qkv + (size_t)DD * DD,     DD, DD);
    k_transpose_f32_f16<<<dim3(DD / 32, DD / 32), tpb_t, 0, stream>>>(Wv, wt_qkv + (size_t)2 * DD * DD, DD, DD);
    k_transpose_f32_f16<<<dim3(DD / 32, DD / 32), tpb_t, 0, stream>>>(Wo, wt_o,                  DD, DD);
    k_transpose_f32_f16<<<dim3(2 * DD / 32, DD / 32), tpb_t, 0, stream>>>(W1, wt_1, DD, 2 * DD);
    k_transpose_f32_f16<<<dim3(DD / 32, 2 * DD / 32), tpb_t, 0, stream>>>(W2, wt_2, 2 * DD, DD);

    // LN1
    k_layernorm_f16<<<MTOT, 256, 0, stream>>>(x, ln1g, ln1b, h1);

    // fused QKV projection: [4096 x 1024] * [1024 x 3072] -> scatter [3][B,H,T,HD]
    k_gemm_f16_wmma<<<dim3(3 * DD / BN, MTOT / BM), 256, 0, stream>>>(
        h1, wt_qkv, MTOT, 3 * DD, DD, /*mode=*/0, nullptr, nullptr, nullptr, qkv);

    // causal flash attention -> ctx [B, T, D]
    k_flash_attn<<<dim3(TT / 128, HH, BBATCH), 256, 0, stream>>>(
        qkv, qkv + (size_t)BBATCH * TT * DD, qkv + (size_t)2 * BBATCH * TT * DD, ctxb);

    // Wo projection + residual -> x1 (f32)
    k_gemm_f16_wmma<<<dim3(DD / BN, MTOT / BM), 256, 0, stream>>>(
        ctxb, wt_o, MTOT, DD, DD, /*mode=*/1, nullptr, x, x1, nullptr);

    // LN2
    k_layernorm_f16<<<MTOT, 256, 0, stream>>>(x1, ln2g, ln2b, h2);

    // MLP up: bias + exact GELU -> f16
    k_gemm_f16_wmma<<<dim3(2 * DD / BN, MTOT / BM), 256, 0, stream>>>(
        h2, wt_1, MTOT, 2 * DD, DD, /*mode=*/2, b1, nullptr, nullptr, ffn);

    // MLP down: bias + residual -> d_out (f32)
    k_gemm_f16_wmma<<<dim3(DD / BN, MTOT / BM), 256, 0, stream>>>(
        ffn, wt_2, MTOT, DD, 2 * DD, /*mode=*/3, b2, x1, out, nullptr);
}
